// NVFP4FakeQuantizedMoE_21749714387162
// MI455X (gfx1250) — matrix operand order repaired and checked by reading, not compiled, not run on hardware
//
#include <hip/hip_runtime.h>

// ---------------------------------------------------------------------------
// NVFP4 fake-quantized MoE forward for MI455X (gfx1250, wave32, WMMA bf16)
// ---------------------------------------------------------------------------
typedef __bf16 bf16_t;
typedef bf16_t v16bf __attribute__((ext_vector_type(16)));
typedef bf16_t v8bf  __attribute__((ext_vector_type(8)));
typedef float  v8f   __attribute__((ext_vector_type(8)));

#define T_TOK 2048
#define E_EXP 8
#define H_DIM 1024
#define I_DIM 768
#define K_TOP 2
#define NROWS (T_TOK * K_TOP)               // 4096 routed rows
#define MAX_TILES 264                        // ceil((4096 + 8*15)/16)

// ---------------- workspace layout (bytes) ----------------
// [0..255]   control u32s: amax_x(0), amax_w1[8](4), amax_w2[8](36),
//            amax_act(68), counts[8](72), cursor[8](104), offs[9](136),
//            numTiles(172)
static constexpr size_t OFF_SORTED = 256;                                   // 4224 ints
static constexpr size_t OFF_TILEE  = OFF_SORTED + (size_t)MAX_TILES*16*4;   // 264 ints
static constexpr size_t OFF_XQ     = 18688;                                 // bf16 [2048,1024]
static constexpr size_t OFF_W1Q    = OFF_XQ  + (size_t)T_TOK*H_DIM*2;       // bf16 [8,1536,1024]
static constexpr size_t OFF_W2Q    = OFF_W1Q + (size_t)E_EXP*2*I_DIM*H_DIM*2;// bf16 [8,1024,768]
static constexpr size_t OFF_ACT    = OFF_W2Q + (size_t)E_EXP*H_DIM*I_DIM*2; // bf16 [4096,768]
static constexpr size_t OFF_ACTQ   = OFF_ACT + (size_t)NROWS*I_DIM*2;       // bf16 [4096,768]
static constexpr size_t OFF_G2     = OFF_ACTQ+ (size_t)NROWS*I_DIM*2;       // f32  [4096,1024]

// ---------------- helpers ----------------
__device__ __forceinline__ float bf16r(float x) { return (float)(bf16_t)x; }

// round-to-nearest-even quantize of non-negative f32 to FP8 E4M3 (fn, sat 448)
__device__ __forceinline__ float e4m3_q(float x) {
    if (!(x > 0.f)) return 0.f;
    if (x >= 448.f) return 448.f;
    union { float f; unsigned u; } v; v.f = x;
    int ex = (int)((v.u >> 23) & 0xFF) - 127;
    if (ex < -6) {                       // subnormal region: step 2^-9
        return rintf(x * 512.f) * (1.f / 512.f);
    }
    unsigned keep = v.u & 0xFFF00000u;   // keep 3 mantissa bits
    unsigned rem  = v.u & 0x000FFFFFu;
    if (rem > 0x00080000u || (rem == 0x00080000u && (keep & 0x00100000u)))
        keep += 0x00100000u;
    v.u = keep;
    return fminf(v.f, 448.f);
}

// E2M1 magnitude via searchsorted(left) on boundaries
__device__ __forceinline__ float e2m1_mag(float a) {
    const float vals[8] = {0.f, 0.5f, 1.f, 1.5f, 2.f, 3.f, 4.f, 6.f};
    int idx = (a > 0.25f) + (a > 0.75f) + (a > 1.25f) + (a > 1.75f) +
              (a > 2.5f)  + (a > 3.5f)  + (a > 5.0f);
    return vals[idx];
}

// ---------------- kernels ----------------
__global__ void k_init(unsigned* ctrl) {
    if (threadIdx.x < 64) ctrl[threadIdx.x] = 0u;
}

// global amax of |bf16(x)| per group (positive float bits compare as uint)
__global__ void k_amax(const float* __restrict__ x, size_t per_group,
                       unsigned* __restrict__ out) {
    int g = blockIdx.y;
    const float* p = x + per_group * (size_t)g;
    float m = 0.f;
    for (size_t i = (size_t)blockIdx.x * blockDim.x + threadIdx.x;
         i < per_group; i += (size_t)gridDim.x * blockDim.x)
        m = fmaxf(m, fabsf(bf16r(p[i])));
    for (int o = 16; o > 0; o >>= 1) m = fmaxf(m, __shfl_xor(m, o, 32));
    if ((threadIdx.x & 31) == 0) atomicMax(&out[g], __float_as_uint(m));
}

// block-16 NVFP4 core-quantize: store xq*block_sf as bf16 (gsf folded out)
template <typename Tin>
__global__ void k_quant_core(const Tin* __restrict__ x, bf16_t* __restrict__ q,
                             const unsigned* __restrict__ amax,
                             size_t per_group, int groups) {
    size_t bpg = per_group / 16;
    size_t nblk = bpg * (size_t)groups;
    for (size_t b = (size_t)blockIdx.x * blockDim.x + threadIdx.x;
         b < nblk; b += (size_t)gridDim.x * blockDim.x) {
        int g = (int)(b / bpg);
        float gsf = 2688.f / __uint_as_float(amax[g]);   // 448*6/amax
        const Tin* p = x + b * 16;
        float xs[16]; float bm = 0.f;
#pragma unroll
        for (int i = 0; i < 16; ++i) {
            xs[i] = bf16r((float)p[i]) * gsf;
            bm = fmaxf(bm, fabsf(xs[i]));
        }
        float bsf = e4m3_q(bm * (1.f / 6.f));
        float inv = 1.f / fmaxf(bsf, 0.015625f);
        bf16_t* o = q + b * 16;
#pragma unroll
        for (int i = 0; i < 16; ++i) {
            float xn = xs[i] * inv;
            float s = (xn < 0.f) ? -1.f : (xn > 0.f ? 1.f : 0.f);
            o[i] = (bf16_t)(s * e2m1_mag(fabsf(xn)) * bsf);  // exact in bf16
        }
    }
}

__global__ void k_count(const int* __restrict__ ri, unsigned* __restrict__ counts) {
    int r = blockIdx.x * blockDim.x + threadIdx.x;
    if (r < NROWS) {
        int e = min(max(ri[r], 0), E_EXP - 1);
        atomicAdd(&counts[e], 1u);
    }
}

__global__ void k_setup(const unsigned* __restrict__ counts,
                        unsigned* __restrict__ cursor, unsigned* __restrict__ offs,
                        unsigned* __restrict__ numTiles, int* __restrict__ sorted,
                        int* __restrict__ tilee) {
    __shared__ unsigned so[9];
    if (threadIdx.x == 0) {
        unsigned off = 0;
        for (int e = 0; e < E_EXP; ++e) {
            so[e] = off;
            off += (counts[e] + 15u) & ~15u;   // pad segment to 16 rows
        }
        so[8] = off;
        for (int e = 0; e < 9; ++e) offs[e] = so[e];
        for (int e = 0; e < E_EXP; ++e) cursor[e] = so[e];
        *numTiles = off / 16u;
    }
    __syncthreads();
    for (int i = threadIdx.x; i < MAX_TILES * 16; i += blockDim.x) sorted[i] = -1;
    for (int t = threadIdx.x; t < MAX_TILES; t += blockDim.x) {
        unsigned base = (unsigned)t * 16u; int e = E_EXP - 1;
        for (int j = 0; j < E_EXP; ++j)
            if (base >= so[j] && base < so[j + 1]) { e = j; break; }
        tilee[t] = e;
    }
}

__global__ void k_scatter(const int* __restrict__ ri, unsigned* __restrict__ cursor,
                          int* __restrict__ sorted) {
    int r = blockIdx.x * blockDim.x + threadIdx.x;
    if (r < NROWS) {
        int e = min(max(ri[r], 0), E_EXP - 1);
        unsigned p = atomicAdd(&cursor[e], 1u);
        sorted[p] = r;
    }
}

// branch-free A fragment load (16x32 bf16, contiguous k = 16*(lane/16) + s)
__device__ __forceinline__ v16bf load_a_frag(const bf16_t* __restrict__ arow,
                                             int k, int khalfA) {
    v8bf lo = *(const v8bf*)(arow + k + khalfA);
    v8bf hi = *(const v8bf*)(arow + k + 8 + khalfA);
    v16bf a;
#pragma unroll
    for (int i = 0; i < 8; ++i) { a[i] = lo[i]; a[i + 8] = hi[i]; }
    return a;
}

// GEMM1 + SiLU fusion: each wave computes value tile (cols cb..cb+15) and
// gate tile (cols I+cb..I+cb+15), writes bf16 act + act amax.
// Sentinel rows redirect their A pointer to the tile's first (always valid)
// row so the k-loop has no divergence; their output is zeroed in epilogue.
__global__ __launch_bounds__(256)
void k_gemm1(const bf16_t* __restrict__ xq, const bf16_t* __restrict__ w1q,
             const int* __restrict__ sorted, const int* __restrict__ tilee,
             const unsigned* __restrict__ numTiles,
             const unsigned* __restrict__ amax_x, const unsigned* __restrict__ amax_w1,
             bf16_t* __restrict__ act, unsigned* __restrict__ amax_act) {
    int tile = blockIdx.x;
    if ((unsigned)tile >= *numTiles) return;
    int e    = tilee[tile];
    int wave = threadIdx.x >> 5;
    int lane = threadIdx.x & 31;
    int cb   = (blockIdx.y * 8 + wave) * 16;          // [0, I)

    __shared__ int rows[16];
    if (threadIdx.x < 16) rows[threadIdx.x] = sorted[tile * 16 + threadIdx.x];
    __syncthreads();

    int m = lane & 15;
    int khalfA = (lane >> 4) * 16;
    int khalfB = (lane >> 4) * 16;
    int rr    = rows[m];
    int rsafe = (rr >= 0) ? rr : rows[0];             // rows[0] always valid
    const bf16_t* arow   = xq + (size_t)(rsafe / K_TOP) * H_DIM;
    const bf16_t* wbase  = w1q + (size_t)e * (2 * I_DIM) * H_DIM;
    const bf16_t* brow_v = wbase + (size_t)(cb + m) * H_DIM;          // value cols
    const bf16_t* brow_g = wbase + (size_t)(I_DIM + cb + m) * H_DIM;  // gate cols

    v8f cv = {0.f, 0.f, 0.f, 0.f, 0.f, 0.f, 0.f, 0.f};
    v8f cg = cv;
    for (int k = 0; k < H_DIM; k += 32) {
        __builtin_prefetch((const void*)(brow_v + k + 128), 0, 1);
        __builtin_prefetch((const void*)(brow_g + k + 128), 0, 1);
        v16bf a  = load_a_frag(arow, k, khalfA);
        v16bf bv = *(const v16bf*)(brow_v + k + khalfB);
        v16bf bg = *(const v16bf*)(brow_g + k + khalfB);
        cv = __builtin_amdgcn_wmma_f32_16x16x32_bf16(false, a, false, bv,
                                                     (short)0, cv, false, false);
        cg = __builtin_amdgcn_wmma_f32_16x16x32_bf16(false, a, false, bg,
                                                     (short)0, cg, false, false);
    }
    // dequant scale: 1/(gsf_x * gsf_w1[e]) = amax_x*amax_w1/(2688^2)
    float s = __uint_as_float(*amax_x) * __uint_as_float(amax_w1[e]) *
              (1.f / (2688.f * 2688.f));
    float lm = 0.f;
    int n = lane & 15;
#pragma unroll
    for (int v = 0; v < 8; ++v) {
        int mrow = v + ((lane >> 4) << 3);
        int rrr  = rows[mrow];
        float val = cv[v] * s;
        float gat = cg[v] * s;
        float a1  = gat / (1.f + __expf(-gat)) * val;     // silu(gate)*value
        if (rrr < 0) a1 = 0.f;                            // kill sentinel rows
        bf16_t ab = (bf16_t)a1;
        lm = fmaxf(lm, fabsf((float)ab));
        if (rrr >= 0) act[(size_t)rrr * I_DIM + cb + n] = ab;
    }
    for (int o = 16; o > 0; o >>= 1) lm = fmaxf(lm, __shfl_xor(lm, o, 32));
    if (lane == 0) atomicMax(amax_act, __float_as_uint(lm));
}

// GEMM2 + per-row top-k weight; writes weighted g2 rows (f32)
__global__ __launch_bounds__(256)
void k_gemm2(const bf16_t* __restrict__ actq, const bf16_t* __restrict__ w2q,
             const int* __restrict__ sorted, const int* __restrict__ tilee,
             const unsigned* __restrict__ numTiles,
             const unsigned* __restrict__ amax_act, const unsigned* __restrict__ amax_w2,
             const int* __restrict__ ri, const float* __restrict__ rw,
             float* __restrict__ g2) {
    int tile = blockIdx.x;
    if ((unsigned)tile >= *numTiles) return;
    int e    = tilee[tile];
    int wave = threadIdx.x >> 5;
    int lane = threadIdx.x & 31;
    int cb   = (blockIdx.y * 8 + wave) * 16;          // [0, H)

    __shared__ int rows[16];
    if (threadIdx.x < 16) rows[threadIdx.x] = sorted[tile * 16 + threadIdx.x];
    __syncthreads();

    int m = lane & 15;
    int khalfA = (lane >> 4) * 16;
    int khalfB = (lane >> 4) * 16;
    int rr    = rows[m];
    int rsafe = (rr >= 0) ? rr : rows[0];             // rows[0] always valid
    const bf16_t* arow = actq + (size_t)rsafe * I_DIM;
    const bf16_t* brow = w2q + (size_t)e * H_DIM * I_DIM + (size_t)(cb + m) * I_DIM;

    v8f c = {0.f, 0.f, 0.f, 0.f, 0.f, 0.f, 0.f, 0.f};
    for (int k = 0; k < I_DIM; k += 32) {
        __builtin_prefetch((const void*)(brow + k + 128), 0, 1);
        v16bf a = load_a_frag(arow, k, khalfA);
        v16bf b = *(const v16bf*)(brow + k + khalfB);
        c = __builtin_amdgcn_wmma_f32_16x16x32_bf16(false, a, false, b,
                                                    (short)0, c, false, false);
    }
    float s = __uint_as_float(*amax_act) * __uint_as_float(amax_w2[e]) *
              (1.f / (2688.f * 2688.f));
    int n = lane & 15;
#pragma unroll
    for (int v = 0; v < 8; ++v) {
        int mrow = v + ((lane >> 4) << 3);
        int rrr  = rows[mrow];
        if (rrr >= 0) {
            int t = rrr / K_TOP, kk = rrr % K_TOP;
            int raw = ri[t * K_TOP + kk];
            int ec  = min(max(raw, 0), E_EXP - 1);
            float tw = (raw >= E_EXP) ? 0.f : rw[t * E_EXP + ec];
            g2[(size_t)rrr * H_DIM + cb + n] = c[v] * s * tw;
        }
    }
}

__global__ void k_combine(const float* __restrict__ g2, float* __restrict__ out) {
    int i = blockIdx.x * blockDim.x + threadIdx.x;
    if (i < T_TOK * H_DIM) {
        int t = i / H_DIM, h = i - t * H_DIM;
        out[i] = g2[(size_t)(t * 2) * H_DIM + h] +
                 g2[(size_t)(t * 2 + 1) * H_DIM + h];
    }
}

// ---------------------------------------------------------------------------
extern "C" void kernel_launch(void* const* d_in, const int* in_sizes, int n_in,
                              void* d_out, int out_size, void* d_ws, size_t ws_size,
                              hipStream_t stream) {
    (void)in_sizes; (void)n_in; (void)out_size; (void)ws_size;
    const float* hs = (const float*)d_in[0];   // (2048,1024)
    const float* rw = (const float*)d_in[1];   // (2048,8)
    const float* w1 = (const float*)d_in[2];   // (8,1536,1024)
    const float* w2 = (const float*)d_in[3];   // (8,1024,768)
    const int*   ri = (const int*)d_in[4];     // (2048,2)
    float* out = (float*)d_out;

    char* ws = (char*)d_ws;
    unsigned* ctrl     = (unsigned*)ws;
    unsigned* amax_x   = (unsigned*)(ws + 0);
    unsigned* amax_w1  = (unsigned*)(ws + 4);
    unsigned* amax_w2  = (unsigned*)(ws + 36);
    unsigned* amax_act = (unsigned*)(ws + 68);
    unsigned* counts   = (unsigned*)(ws + 72);
    unsigned* cursor   = (unsigned*)(ws + 104);
    unsigned* offs     = (unsigned*)(ws + 136);
    unsigned* numTiles = (unsigned*)(ws + 172);
    int*    sorted = (int*)(ws + OFF_SORTED);
    int*    tilee  = (int*)(ws + OFF_TILEE);
    bf16_t* xq     = (bf16_t*)(ws + OFF_XQ);
    bf16_t* w1q    = (bf16_t*)(ws + OFF_W1Q);
    bf16_t* w2q    = (bf16_t*)(ws + OFF_W2Q);
    bf16_t* act    = (bf16_t*)(ws + OFF_ACT);
    bf16_t* actq   = (bf16_t*)(ws + OFF_ACTQ);
    float*  g2     = (float*)(ws + OFF_G2);

    const size_t nx  = (size_t)T_TOK * H_DIM;            // 2,097,152
    const size_t nw1 = (size_t)2 * I_DIM * H_DIM;        // per-expert 1,572,864
    const size_t nw2 = (size_t)H_DIM * I_DIM;            // per-expert   786,432
    const size_t na  = (size_t)NROWS * I_DIM;            // 3,145,728

    k_init<<<1, 64, 0, stream>>>(ctrl);

    k_amax<<<dim3(256, 1), 256, 0, stream>>>(hs, nx, amax_x);
    k_amax<<<dim3(128, E_EXP), 256, 0, stream>>>(w1, nw1, amax_w1);
    k_amax<<<dim3(64, E_EXP), 256, 0, stream>>>(w2, nw2, amax_w2);

    k_quant_core<float><<<(unsigned)((nx / 16 + 255) / 256), 256, 0, stream>>>(
        hs, xq, amax_x, nx, 1);
    k_quant_core<float><<<2048, 256, 0, stream>>>(w1, w1q, amax_w1, nw1, E_EXP);
    k_quant_core<float><<<1024, 256, 0, stream>>>(w2, w2q, amax_w2, nw2, E_EXP);

    k_count<<<(NROWS + 255) / 256, 256, 0, stream>>>(ri, counts);
    k_setup<<<1, 256, 0, stream>>>(counts, cursor, offs, numTiles, sorted, tilee);
    k_scatter<<<(NROWS + 255) / 256, 256, 0, stream>>>(ri, cursor, sorted);

    // GEMM1 (+SiLU, act amax): 48 value-col tiles -> grid.y = 6 (8 waves/blk)
    k_gemm1<<<dim3(MAX_TILES, I_DIM / (16 * 8)), 256, 0, stream>>>(
        xq, w1q, sorted, tilee, numTiles, amax_x, amax_w1, act, amax_act);

    k_quant_core<bf16_t><<<(unsigned)((na / 16 + 255) / 256), 256, 0, stream>>>(
        act, actq, amax_act, na, 1);

    // GEMM2 (+topk weight): 64 col tiles -> grid.y = 8
    k_gemm2<<<dim3(MAX_TILES, H_DIM / (16 * 8)), 256, 0, stream>>>(
        actq, w2q, sorted, tilee, numTiles, amax_act, amax_w2, ri, rw, g2);

    k_combine<<<(T_TOK * H_DIM + 255) / 256, 256, 0, stream>>>(g2, out);
}
